// Block_21672404976291
// MI455X (gfx1250) — matrix operand order repaired and checked by reading, not compiled, not run on hardware
//
#include <hip/hip_runtime.h>

typedef __attribute__((ext_vector_type(16))) __bf16 v16bf;
typedef __attribute__((ext_vector_type(8)))  __bf16 v8bf;
typedef __attribute__((ext_vector_type(8)))  float  v8f;
typedef __attribute__((ext_vector_type(4)))  int    v4i;

static constexpr int C   = 768;
static constexpr int HN  = 12;
static constexpr int HD  = 64;
static constexpr int Bsz = 4;
static constexpr int T   = 2048;
static constexpr int BT  = Bsz * T;   // 8192 tokens
static constexpr int C3  = 3 * C;     // 2304
static constexpr int C4  = 4 * C;     // 3072

#define AS1 __attribute__((address_space(1)))
#define AS3 __attribute__((address_space(3)))

#if defined(__has_builtin)
#if __has_builtin(__builtin_amdgcn_global_load_async_to_lds_b128) && \
    __has_builtin(__builtin_amdgcn_s_wait_asynccnt)
#define USE_ASYNC_LDS 1
#endif
#endif
#ifndef USE_ASYNC_LDS
#define USE_ASYNC_LDS 0
#endif

// ---------- helpers ----------

__device__ __forceinline__ __bf16 f2bf(float x) {
  unsigned u = __float_as_uint(x);
  unsigned r = (u + 0x7FFFu + ((u >> 16) & 1u)) >> 16;   // round-to-nearest-even
  unsigned short s = (unsigned short)r;
  return __builtin_bit_cast(__bf16, s);
}

__device__ __forceinline__ float gelu_tanh(float x) {
  const float c = 0.7978845608028654f;   // sqrt(2/pi)
  float x3 = x * x * x;
  return 0.5f * x * (1.0f + tanhf(c * (x + 0.044715f * x3)));
}

__device__ __forceinline__ v8f wmma_bf16(v16bf a, v16bf b, v8f c) {
  return __builtin_amdgcn_wmma_f32_16x16x32_bf16(false, a, false, b,
                                                 (short)0, c, false, false);
}

// A-fragment (16x32 bf16): lane -> row m, K = {g*8..g*8+7} U {16+g*8..16+g*8+7}
__device__ __forceinline__ v16bf load_fragA(const __bf16* base, int row, int ld,
                                            int k0, int g) {
  const __bf16* p = base + (size_t)row * ld + k0 + g * 8;
  union { v16bf v; v8bf h[2]; } u;
  u.h[0] = *reinterpret_cast<const v8bf*>(p);
  u.h[1] = *reinterpret_cast<const v8bf*>(p + 16);
  return u.v;
}

// B-fragment (32x16 bf16) from B^T row-major: lane -> column n, K = {g*8..g*8+7} U {16+g*8..16+g*8+7}
__device__ __forceinline__ v16bf load_fragB(const __bf16* base, int row, int ld,
                                            int k0, int g) {
  const __bf16* p = base + (size_t)row * ld + k0 + g * 8;
  union { v16bf v; v8bf h[2]; } u;
  u.h[0] = *reinterpret_cast<const v8bf*>(p);
  u.h[1] = *reinterpret_cast<const v8bf*>(p + 16);
  return u.v;
}

// ---------- weight cast + transpose: W[K][N] fp32 -> Wt[N][K] bf16 ----------

__global__ void cast_transpose(const float* __restrict__ W,
                               __bf16* __restrict__ Wt, int Krows, int Ncols) {
  int idx = blockIdx.x * 256 + threadIdx.x;
  if (idx >= Krows * Ncols) return;
  int k = idx / Ncols;
  int n = idx - k * Ncols;
  Wt[(size_t)n * Krows + k] = f2bf(W[idx]);
}

// ---------- LayerNorm (fp32 in -> bf16 out), one block per token ----------

__global__ __launch_bounds__(256) void layernorm_to_bf16(
    const float* __restrict__ x, const float* __restrict__ gam,
    const float* __restrict__ bet, __bf16* __restrict__ out) {
  const int row = blockIdx.x;
  const float* xr = x + (size_t)row * C;
  float v[3];
  float s = 0.f, q = 0.f;
#pragma unroll
  for (int i = 0; i < 3; ++i) {
    v[i] = xr[threadIdx.x + i * 256];
    s += v[i];
    q += v[i] * v[i];
  }
#pragma unroll
  for (int off = 16; off > 0; off >>= 1) {
    s += __shfl_xor(s, off, 32);
    q += __shfl_xor(q, off, 32);
  }
  __shared__ float rs[8], rq[8];
  const int wave = threadIdx.x >> 5;
  if ((threadIdx.x & 31) == 0) { rs[wave] = s; rq[wave] = q; }
  __syncthreads();
  float ts = 0.f, tq = 0.f;
#pragma unroll
  for (int i = 0; i < 8; ++i) { ts += rs[i]; tq += rq[i]; }
  const float mu   = ts * (1.0f / C);
  const float var  = tq * (1.0f / C) - mu * mu;
  const float rsig = rsqrtf(var + 1e-5f);
#pragma unroll
  for (int i = 0; i < 3; ++i) {
    int cidx = threadIdx.x + i * 256;
    out[(size_t)row * C + cidx] = f2bf((v[i] - mu) * rsig * gam[cidx] + bet[cidx]);
  }
}

// ---------- Tiled bf16 WMMA GEMM: C[M,N] = A[M,K] * (Wt[N,K])^T ----------
// Block tile 256(M) x 64(N), 8 waves, each wave a 32x64 tile (8 WMMAs/K-step).
// MODE 0: qkv scatter (out0=Q scaled 1/8, out1=K, out2=V^T)
// MODE 1: bias + residual -> fp32 out0
// MODE 2: bias + GELU    -> bf16 out0

template <int MODE>
__global__ __launch_bounds__(256) void gemm_bf16_wmma(
    const __bf16* __restrict__ A, const __bf16* __restrict__ Wt,
    const float* __restrict__ bias, const float* __restrict__ resid,
    void* __restrict__ out0, void* __restrict__ out1, void* __restrict__ out2,
    int M, int N, int K) {
  __shared__ __bf16 Bs[64 * 32];
  const int lane = threadIdx.x & 31;
  const int wave = threadIdx.x >> 5;
  const int g    = lane >> 4;
  const int nl   = lane & 15;
  const int n0   = blockIdx.x * 64;
  const int m0   = blockIdx.y * 256 + wave * 32;

  v8f acc[2][4];
#pragma unroll
  for (int mi = 0; mi < 2; ++mi)
#pragma unroll
    for (int i = 0; i < 4; ++i)
#pragma unroll
      for (int r = 0; r < 8; ++r) acc[mi][i][r] = 0.0f;

  const int srow = threadIdx.x >> 2;   // staging row  (0..63)
  const int sch  = threadIdx.x & 3;    // staging chunk (0..3), 8 bf16 each

  for (int k0 = 0; k0 < K; k0 += 32) {
    // stage B^T tile [64 n][32 k] into LDS (4 KB), one b128 per thread
    const __bf16* src = Wt + (size_t)(n0 + srow) * K + k0 + sch * 8;
#if USE_ASYNC_LDS
    __builtin_amdgcn_global_load_async_to_lds_b128(
        (AS1 v4i*)src, (AS3 v4i*)&Bs[srow * 32 + sch * 8], 0, 0);
#else
    *reinterpret_cast<v8bf*>(&Bs[srow * 32 + sch * 8]) =
        *reinterpret_cast<const v8bf*>(src);
#endif
    if (k0 + 32 < K)
      __builtin_prefetch(src + 32, 0, 1);       // next K tile of weights
#if USE_ASYNC_LDS
    __builtin_amdgcn_s_wait_asynccnt(0);
#endif
    __syncthreads();

    v16bf a0 = load_fragA(A, m0 + nl,      K, k0, g);
    v16bf a1 = load_fragA(A, m0 + 16 + nl, K, k0, g);
#pragma unroll
    for (int nf = 0; nf < 4; ++nf) {
      v16bf bfrag = load_fragB(Bs, nf * 16 + nl, 32, 0, g);
      acc[0][nf] = wmma_bf16(a0, bfrag, acc[0][nf]);
      acc[1][nf] = wmma_bf16(a1, bfrag, acc[1][nf]);
    }
    __syncthreads();
  }

#pragma unroll
  for (int mi = 0; mi < 2; ++mi) {
#pragma unroll
    for (int nf = 0; nf < 4; ++nf) {
      int col = n0 + nf * 16 + nl;
      float bc = bias[col];
#pragma unroll
      for (int r = 0; r < 8; ++r) {
        int row = m0 + mi * 16 + g * 8 + r;
        float v = acc[mi][nf][r] + bc;
        if (MODE == 0) {
          int sec  = col / C;
          int c    = col - sec * C;
          int head = c >> 6, d = c & 63;
          int b = row >> 11, t = row & (T - 1);
          size_t qk = (((size_t)(b * HN + head) * T) + t) * HD + d;
          if (sec == 0) {
            ((__bf16*)out0)[qk] = f2bf(v * 0.125f);     // 1/sqrt(64) folded in
          } else if (sec == 1) {
            ((__bf16*)out1)[qk] = f2bf(v);
          } else {
            size_t vt = ((size_t)(b * HN + head) * HD + d) * T + t;  // V^T [D][T]
            ((__bf16*)out2)[vt] = f2bf(v);
          }
        } else if (MODE == 1) {
          ((float*)out0)[(size_t)row * N + col] = v + resid[(size_t)row * N + col];
        } else {
          ((__bf16*)out0)[(size_t)row * N + col] = f2bf(gelu_tanh(v));
        }
      }
    }
  }
}

// ---------- Flash attention: 1 wave per (b,h, 16-query tile) ----------

__global__ __launch_bounds__(32) void flash_attn(
    const __bf16* __restrict__ Q, const __bf16* __restrict__ Kk,
    const __bf16* __restrict__ Vt, __bf16* __restrict__ Y) {
  __shared__ __bf16 Pt[16 * 32];
  const int lane = threadIdx.x;
  const int g = lane >> 4, nl = lane & 15;
  const int bh = blockIdx.y;
  const int m0 = blockIdx.x * 16;
  const __bf16* Qh = Q  + (size_t)bh * T * HD;
  const __bf16* Kh = Kk + (size_t)bh * T * HD;
  const __bf16* Vh = Vt + (size_t)bh * HD * T;   // [D][T]

  v16bf aq0 = load_fragA(Qh, m0 + nl, HD, 0, g);
  v16bf aq1 = load_fragA(Qh, m0 + nl, HD, 32, g);

  v8f o[4];
  float mrow[8], lrow[8];
#pragma unroll
  for (int dt = 0; dt < 4; ++dt)
#pragma unroll
    for (int r = 0; r < 8; ++r) o[dt][r] = 0.0f;
#pragma unroll
  for (int r = 0; r < 8; ++r) { mrow[r] = -1e30f; lrow[r] = 0.0f; }

  const int kend = m0 + 16;                  // causal: keys <= m0+15
  for (int kb = 0; kb < kend; kb += 32) {
    v8f s0, s1;
#pragma unroll
    for (int r = 0; r < 8; ++r) { s0[r] = 0.f; s1[r] = 0.f; }
    {
      v16bf b0 = load_fragB(Kh, kb + nl, HD, 0, g);
      v16bf b1 = load_fragB(Kh, kb + nl, HD, 32, g);
      s0 = wmma_bf16(aq0, b0, s0);
      s0 = wmma_bf16(aq1, b1, s0);
    }
    {
      v16bf b0 = load_fragB(Kh, kb + 16 + nl, HD, 0, g);
      v16bf b1 = load_fragB(Kh, kb + 16 + nl, HD, 32, g);
      s1 = wmma_bf16(aq0, b0, s1);
      s1 = wmma_bf16(aq1, b1, s1);
    }
    // causal mask
#pragma unroll
    for (int r = 0; r < 8; ++r) {
      int qrow = m0 + g * 8 + r;
      if (kb + nl > qrow)      s0[r] = -1e30f;
      if (kb + 16 + nl > qrow) s1[r] = -1e30f;
    }
    // online softmax (rows live in 16-lane groups; xor 1,2,4,8 stays in-group)
    float scale[8];
#pragma unroll
    for (int r = 0; r < 8; ++r) {
      float mx = fmaxf(s0[r], s1[r]);
#pragma unroll
      for (int off = 1; off < 16; off <<= 1) mx = fmaxf(mx, __shfl_xor(mx, off, 32));
      float nm = fmaxf(mrow[r], mx);
      scale[r] = __expf(mrow[r] - nm);
      mrow[r]  = nm;
      float p0 = __expf(s0[r] - nm);
      float p1 = __expf(s1[r] - nm);
      s0[r] = p0; s1[r] = p1;
      float rsum = p0 + p1;
#pragma unroll
      for (int off = 1; off < 16; off <<= 1) rsum += __shfl_xor(rsum, off, 32);
      lrow[r] = lrow[r] * scale[r] + rsum;
    }
#pragma unroll
    for (int dt = 0; dt < 4; ++dt)
#pragma unroll
      for (int r = 0; r < 8; ++r) o[dt][r] *= scale[r];
    // P (C-layout) -> LDS -> A-layout fragment
#pragma unroll
    for (int r = 0; r < 8; ++r) {
      Pt[(g * 8 + r) * 32 + nl]      = f2bf(s0[r]);
      Pt[(g * 8 + r) * 32 + 16 + nl] = f2bf(s1[r]);
    }
    __syncthreads();
    v16bf pf = load_fragA(Pt, nl, 32, 0, g);
#pragma unroll
    for (int dt = 0; dt < 4; ++dt) {
      v16bf vf = load_fragB(Vh, dt * 16 + nl, T, kb, g);
      o[dt] = wmma_bf16(pf, vf, o[dt]);
    }
    __syncthreads();
  }
  // normalize and write y [tokens][C], col = head*64 + d
  const int b = bh / HN, head = bh - b * HN;
#pragma unroll
  for (int r = 0; r < 8; ++r) {
    float inv = 1.0f / lrow[r];
    int tok = b * T + m0 + g * 8 + r;
#pragma unroll
    for (int dt = 0; dt < 4; ++dt) {
      int colc = head * HD + dt * 16 + nl;
      Y[(size_t)tok * C + colc] = f2bf(o[dt][r] * inv);
    }
  }
}

// ---------- launch ----------

extern "C" void kernel_launch(void* const* d_in, const int* in_sizes, int n_in,
                              void* d_out, int out_size, void* d_ws, size_t ws_size,
                              hipStream_t stream) {
  (void)in_sizes; (void)n_in; (void)out_size; (void)ws_size;
  const float* x      = (const float*)d_in[0];
  const float* ln1_g  = (const float*)d_in[1];
  const float* ln1_b  = (const float*)d_in[2];
  const float* W_attn = (const float*)d_in[3];
  const float* b_attn = (const float*)d_in[4];
  const float* W_proj = (const float*)d_in[5];
  const float* b_proj = (const float*)d_in[6];
  const float* ln2_g  = (const float*)d_in[7];
  const float* ln2_b  = (const float*)d_in[8];
  const float* W_fc   = (const float*)d_in[9];
  const float* b_fc   = (const float*)d_in[10];
  const float* W_fc2  = (const float*)d_in[11];
  const float* b_fc2  = (const float*)d_in[12];
  float* out = (float*)d_out;

  char* ws = (char*)d_ws;
  size_t off = 0;
  auto take = [&](size_t bytes) -> char* {
    char* p = ws + off;
    off += (bytes + 255) & ~(size_t)255;
    return p;
  };
  __bf16* h1   = (__bf16*)take((size_t)BT * C  * 2);
  __bf16* WtA  = (__bf16*)take((size_t)C3 * C  * 2);
  __bf16* WtP  = (__bf16*)take((size_t)C  * C  * 2);
  __bf16* WtF  = (__bf16*)take((size_t)C4 * C  * 2);
  __bf16* WtF2 = (__bf16*)take((size_t)C  * C4 * 2);
  __bf16* Qb   = (__bf16*)take((size_t)BT * C  * 2);
  __bf16* Kb   = (__bf16*)take((size_t)BT * C  * 2);
  __bf16* Vtb  = (__bf16*)take((size_t)BT * C  * 2);
  __bf16* Yb   = (__bf16*)take((size_t)BT * C  * 2);
  float*  X1   = (float*) take((size_t)BT * C  * 4);
  __bf16* H2   = (__bf16*)take((size_t)BT * C  * 2);
  __bf16* ACT  = (__bf16*)take((size_t)BT * C4 * 2);

  // weights -> bf16, transposed to [N][K]
  cast_transpose<<<dim3((C * C3 + 255) / 256), 256, 0, stream>>>(W_attn, WtA, C, C3);
  cast_transpose<<<dim3((C * C  + 255) / 256), 256, 0, stream>>>(W_proj, WtP, C, C);
  cast_transpose<<<dim3((C * C4 + 255) / 256), 256, 0, stream>>>(W_fc,   WtF, C, C4);
  cast_transpose<<<dim3((C4 * C + 255) / 256), 256, 0, stream>>>(W_fc2, WtF2, C4, C);

  // LN1
  layernorm_to_bf16<<<dim3(BT), 256, 0, stream>>>(x, ln1_g, ln1_b, h1);

  // qkv = h1 @ W_attn + b  -> scatter Q/K/V^T
  gemm_bf16_wmma<0><<<dim3(C3 / 64, BT / 256), 256, 0, stream>>>(
      h1, WtA, b_attn, nullptr, Qb, Kb, Vtb, BT, C3, C);

  // causal flash attention -> y
  flash_attn<<<dim3(T / 16, Bsz * HN), 32, 0, stream>>>(Qb, Kb, Vtb, Yb);

  // x1 = x + y @ W_proj + b
  gemm_bf16_wmma<1><<<dim3(C / 64, BT / 256), 256, 0, stream>>>(
      Yb, WtP, b_proj, x, X1, nullptr, nullptr, BT, C, C);

  // LN2
  layernorm_to_bf16<<<dim3(BT), 256, 0, stream>>>(X1, ln2_g, ln2_b, H2);

  // act = gelu(h2 @ W_fc + b)
  gemm_bf16_wmma<2><<<dim3(C4 / 64, BT / 256), 256, 0, stream>>>(
      H2, WtF, b_fc, nullptr, ACT, nullptr, nullptr, BT, C4, C);

  // out = x1 + act @ W_fc2 + b
  gemm_bf16_wmma<1><<<dim3(C / 64, BT / 256), 256, 0, stream>>>(
      ACT, WtF2, b_fc2, X1, out, nullptr, nullptr, BT, C, C4);
}
